// HyperbolicAttention_45981919871253
// MI455X (gfx1250) — matrix operand order repaired and checked by reading, not compiled, not run on hardware
//
#include <hip/hip_runtime.h>
#include <math.h>

typedef __attribute__((ext_vector_type(16))) __bf16 v16bf;
typedef __attribute__((ext_vector_type(8)))  __bf16 v8bf;
typedef __attribute__((ext_vector_type(4)))  __bf16 v4bf;
typedef __attribute__((ext_vector_type(8)))  float  v8f;

#define WARP     32
#define WAVES    4
#define BLOCK_T  (WARP * WAVES)
#define SEQ      2048
#define DDIM     256
#define ROWS_BLK 64          // 16 query rows per wave
#define KVT      64          // keys per streaming iteration (4 S-tiles)
#define QSTR     264         // bf16 elems; 528 B rows: bank-conflict-free, 16B aligned
#define KSTR     264
#define VSTR     72          // 144 B rows: 36*lane mod 64 all distinct
#define PSTR     72
#define HEPS     1e-6f

// ---- dynamic LDS partition (bytes) ----
#define OFF_Q   0
#define SZ_Q    (ROWS_BLK * QSTR * 2)       // 33792
#define OFF_K   (OFF_Q + SZ_Q)
#define SZ_K    (KVT * KSTR * 2)            // 33792
#define OFF_V   (OFF_K + SZ_K)
#define SZ_V    (DDIM * VSTR * 2)           // 36864
#define OFF_P   (OFF_V + SZ_V)
#define SZ_P    (WAVES * 16 * PSTR * 2)     // 9216
#define OFF_Q2  (OFF_P + SZ_P)
#define OFF_K2  (OFF_Q2 + ROWS_BLK * 4)
#define OFF_KC  (OFF_K2 + ROWS_BLK * 4)
#define SMEM_SZ (OFF_KC + KVT * 4)          // 114432 B < 320 KB/WGP

// log2-domain hyperbolic logit: softmax is invariant under the consistent
// base-2 rescale, and v_log_f32 / v_exp_f32 are natively log2/exp2.
__device__ __forceinline__ float hyp_logit(float s, float q2, float k2, float di) {
    float nd2 = fmaxf(q2 + k2 - 2.0f * s, 0.0f);
    float arg = fmaxf(fmaf(di, nd2, 1.0f), 1.0f + HEPS);
    float l2  = __builtin_amdgcn_logf(arg + __builtin_amdgcn_sqrtf(arg * arg - 1.0f));
    return l2 * (-0.0625f);                  // -(log2 arccosh-arg)/sqrt(256)
}

__device__ __forceinline__ v16bf cat8(v8bf lo, v8bf hi) {
    return __builtin_shufflevector(lo, hi, 0,1,2,3,4,5,6,7,8,9,10,11,12,13,14,15);
}

__device__ __forceinline__ void store_bf16x4(__bf16* p, float a, float b, float c, float d) {
    v4bf v = { (__bf16)a, (__bf16)b, (__bf16)c, (__bf16)d };
    *(v4bf*)p = v;                           // single ds_store_b64
}

__global__ __launch_bounds__(BLOCK_T) void hyp_attn_kernel(
    const float* __restrict__ Qg, const float* __restrict__ Kg,
    const float* __restrict__ Vg, const float* __restrict__ Cg,
    float* __restrict__ Og)
{
    extern __shared__ __align__(16) char smem[];
    __bf16* Qs    = (__bf16*)(smem + OFF_Q);   // [64][QSTR] row-major bf16
    __bf16* Kl    = (__bf16*)(smem + OFF_K);   // [64][KSTR] row-major bf16
    __bf16* Vt    = (__bf16*)(smem + OFF_V);   // [256][VSTR] transposed bf16
    __bf16* Pl    = (__bf16*)(smem + OFF_P);   // per-wave [16][PSTR]
    float*  q2row = (float*)(smem + OFF_Q2);
    float*  k2row = (float*)(smem + OFF_K2);
    float*  k2col = (float*)(smem + OFF_KC);

    const int t    = threadIdx.x;
    const int wave = t >> 5, lane = t & 31;
    const int half = lane >> 4, ln = lane & 15;
    const int b    = blockIdx.y;
    const int q0   = blockIdx.x * ROWS_BLK;
    const float c  = Cg[0];

    const float* Qb = Qg + (size_t)b * SEQ * DDIM;
    const float* Kb = Kg + (size_t)b * SEQ * DDIM;
    const float* Vb = Vg + (size_t)b * SEQ * DDIM;

    // ---- stage Q (bf16) + fp32 row norms |q_i|^2, |k_i|^2 (denom uses k at ROW idx) ----
    if (t < ROWS_BLK) { q2row[t] = 0.0f; k2row[t] = 0.0f; }
    __syncthreads();
    {
        int row = t >> 1, d0 = (t & 1) * 128;
        const float* qp = Qb + (size_t)(q0 + row) * DDIM + d0;
        const float* kp = Kb + (size_t)(q0 + row) * DDIM + d0;
        float sq = 0.0f, sk = 0.0f;
        #pragma unroll 8
        for (int j = 0; j < 128; j += 4) {
            float4 a = *(const float4*)(qp + j);
            float4 e = *(const float4*)(kp + j);
            sq += a.x*a.x + a.y*a.y + a.z*a.z + a.w*a.w;
            sk += e.x*e.x + e.y*e.y + e.z*e.z + e.w*e.w;
            Qs[row * QSTR + d0 + j + 0] = (__bf16)a.x;
            Qs[row * QSTR + d0 + j + 1] = (__bf16)a.y;
            Qs[row * QSTR + d0 + j + 2] = (__bf16)a.z;
            Qs[row * QSTR + d0 + j + 3] = (__bf16)a.w;
        }
        atomicAdd(&q2row[row], sq);
        atomicAdd(&k2row[row], sk);
    }
    __syncthreads();

    // ---- per-row constants (C-layout: VGPR r -> row r / r+8 by lane half) ----
    float rq2[8], rdi[8], mr[8], lr[8];
    #pragma unroll
    for (int r = 0; r < 8; ++r) {
        int row  = wave * 16 + r + half * 8;
        float q2 = q2row[row], k2 = k2row[row];
        float den = fmaxf((1.0f - c * q2) * (1.0f - c * k2), HEPS);
        rq2[r] = q2;
        rdi[r] = 2.0f * c / den;
        mr[r]  = -3.0e38f;
        lr[r]  = 0.0f;
    }
    const v8f vzero = {0.f,0.f,0.f,0.f,0.f,0.f,0.f,0.f};
    v8f o[16];
    #pragma unroll
    for (int i = 0; i < 16; ++i) o[i] = vzero;

    __bf16* Pw = Pl + wave * 16 * PSTR;
    const int arow = (wave * 16 + ln) * QSTR;

    // ================= stream over keys/values, 64 at a time =================
    for (int kv0 = 0; kv0 < SEQ; kv0 += KVT) {
        __syncthreads();                       // previous tile fully consumed
        if (t < KVT) k2col[t] = 0.0f;
        __syncthreads();
        {   // K tile: row-major bf16 + |k_j|^2
            int kv = t >> 1, d0 = (t & 1) * 128;
            const float* kp = Kb + (size_t)(kv0 + kv) * DDIM + d0;
            if (kv0 + KVT < SEQ) __builtin_prefetch(kp + KVT * DDIM, 0, 1);
            float sk = 0.0f;
            #pragma unroll 8
            for (int j = 0; j < 128; j += 4) {
                float4 a = *(const float4*)(kp + j);
                sk += a.x*a.x + a.y*a.y + a.z*a.z + a.w*a.w;
                Kl[kv * KSTR + d0 + j + 0] = (__bf16)a.x;
                Kl[kv * KSTR + d0 + j + 1] = (__bf16)a.y;
                Kl[kv * KSTR + d0 + j + 2] = (__bf16)a.z;
                Kl[kv * KSTR + d0 + j + 3] = (__bf16)a.w;
            }
            atomicAdd(&k2col[kv], sk);
            // V tile: transposed bf16, 4 kv packed per b64 store
            int kv4 = (t & 15) * 4, dv = (t >> 4) * 32;
            const float* vp = Vb + (size_t)(kv0 + kv4) * DDIM + dv;
            if (kv0 + KVT < SEQ) __builtin_prefetch(vp + KVT * DDIM, 0, 1);
            #pragma unroll
            for (int j = 0; j < 32; j += 4) {
                float4 r0 = *(const float4*)(vp + 0 * DDIM + j);
                float4 r1 = *(const float4*)(vp + 1 * DDIM + j);
                float4 r2 = *(const float4*)(vp + 2 * DDIM + j);
                float4 r3 = *(const float4*)(vp + 3 * DDIM + j);
                store_bf16x4(&Vt[(dv + j + 0) * VSTR + kv4], r0.x, r1.x, r2.x, r3.x);
                store_bf16x4(&Vt[(dv + j + 1) * VSTR + kv4], r0.y, r1.y, r2.y, r3.y);
                store_bf16x4(&Vt[(dv + j + 2) * VSTR + kv4], r0.z, r1.z, r2.z, r3.z);
                store_bf16x4(&Vt[(dv + j + 3) * VSTR + kv4], r0.w, r1.w, r2.w, r3.w);
            }
        }
        __syncthreads();

        // ---- S = Q K^T : four 16x16 f32 tiles via 32 bf16 WMMAs ----
        v8f acc[4];
        #pragma unroll
        for (int nt = 0; nt < 4; ++nt) acc[nt] = vzero;
        #pragma unroll
        for (int kb = 0; kb < 8; ++kb) {
            int ca = kb * 32 + half * 8;       // A: lanes 0-15 K=0..7,16..23; 16-31 +8
            const __bf16* ap = &Qs[arow + ca];
            v16bf am = cat8(*(const v8bf*)ap, *(const v8bf*)(ap + 16));
            int cb = kb * 32 + half * 8;       // B: lanes 0-15 K=0..7,16..23; 16-31 +8
            #pragma unroll
            for (int nt = 0; nt < 4; ++nt) {
                const __bf16* bp = &Kl[(nt * 16 + ln) * KSTR + cb];
                v16bf bm = cat8(*(const v8bf*)bp, *(const v8bf*)(bp + 16));
                acc[nt] = __builtin_amdgcn_wmma_f32_16x16x32_bf16(false, am, false, bm,
                                                                  (short)0, acc[nt], false, false);
            }
        }

        // ---- hyperbolic distance + online softmax (log2 domain) ----
        float k2c[4];
        #pragma unroll
        for (int nt = 0; nt < 4; ++nt) k2c[nt] = k2col[nt * 16 + ln];
        #pragma unroll
        for (int r = 0; r < 8; ++r) {
            float lg[4];
            #pragma unroll
            for (int nt = 0; nt < 4; ++nt)
                lg[nt] = hyp_logit(acc[nt][r], rq2[r], k2c[nt], rdi[r]);
            float mx = fmaxf(fmaxf(lg[0], lg[1]), fmaxf(lg[2], lg[3]));
            #pragma unroll
            for (int off = 8; off >= 1; off >>= 1)
                mx = fmaxf(mx, __shfl_xor(mx, off, WARP));   // stays within 16-group
            float nm = fmaxf(mr[r], mx);
            float p[4], ps = 0.0f;
            #pragma unroll
            for (int nt = 0; nt < 4; ++nt) {
                p[nt] = __builtin_amdgcn_exp2f(lg[nt] - nm);
                ps += p[nt];
            }
            #pragma unroll
            for (int off = 8; off >= 1; off >>= 1)
                ps += __shfl_xor(ps, off, WARP);
            float alpha = __builtin_amdgcn_exp2f(mr[r] - nm);
            lr[r] = lr[r] * alpha + ps;
            mr[r] = nm;
            #pragma unroll
            for (int nt = 0; nt < 16; ++nt) o[nt][r] *= alpha;
            int pr = (r + half * 8) * PSTR;
            #pragma unroll
            for (int nt = 0; nt < 4; ++nt)
                Pw[pr + nt * 16 + ln] = (__bf16)p[nt];
        }

        // ---- O += P V : P is 16x64 -> two A-fragments; 32 WMMAs ----
        const __bf16* pp = &Pw[ln * PSTR + half * 8];
        v16bf pa0 = cat8(*(const v8bf*)pp,        *(const v8bf*)(pp + 16));
        v16bf pa1 = cat8(*(const v8bf*)(pp + 32), *(const v8bf*)(pp + 48));
        #pragma unroll
        for (int nt = 0; nt < 16; ++nt) {
            const __bf16* vpb = &Vt[(nt * 16 + ln) * VSTR + half * 8];
            v16bf vb0 = cat8(*(const v8bf*)vpb,        *(const v8bf*)(vpb + 16));
            v16bf vb1 = cat8(*(const v8bf*)(vpb + 32), *(const v8bf*)(vpb + 48));
            o[nt] = __builtin_amdgcn_wmma_f32_16x16x32_bf16(false, pa0, false, vb0,
                                                            (short)0, o[nt], false, false);
            o[nt] = __builtin_amdgcn_wmma_f32_16x16x32_bf16(false, pa1, false, vb1,
                                                            (short)0, o[nt], false, false);
        }
    }

    // ---- normalize and write fp32 output ----
    #pragma unroll
    for (int r = 0; r < 8; ++r) {
        float li = 1.0f / lr[r];
        float* op = Og + ((size_t)b * SEQ + q0 + wave * 16 + r + half * 8) * DDIM;
        #pragma unroll
        for (int nt = 0; nt < 16; ++nt) op[nt * 16 + ln] = o[nt][r] * li;
    }
}

extern "C" void kernel_launch(void* const* d_in, const int* in_sizes, int n_in,
                              void* d_out, int out_size, void* d_ws, size_t ws_size,
                              hipStream_t stream) {
    const float* Q = (const float*)d_in[0];
    const float* K = (const float*)d_in[1];
    const float* V = (const float*)d_in[2];
    const float* C = (const float*)d_in[3];
    float* Out = (float*)d_out;
    int B = in_sizes[0] / (SEQ * DDIM);        // = 8
    (void)hipFuncSetAttribute((const void*)hyp_attn_kernel,
                              hipFuncAttributeMaxDynamicSharedMemorySize, SMEM_SZ);
    dim3 grid(SEQ / ROWS_BLK, B);
    hyp_attn_kernel<<<grid, dim3(BLOCK_T), SMEM_SZ, stream>>>(Q, K, V, C, Out);
}
